// CasualAttention_25864293057003
// MI455X (gfx1250) — matrix operand order repaired and checked by reading, not compiled, not run on hardware
//
#include <hip/hip_runtime.h>

typedef __attribute__((ext_vector_type(16))) __bf16 v16bf;
typedef __attribute__((ext_vector_type(8)))  __bf16 v8bf;
typedef __attribute__((ext_vector_type(8)))  float  v8f;

#define DM  1024
#define SEQ 4096
#define NB  4

static __device__ __forceinline__ v16bf cat8(v8bf lo, v8bf hi) {
  return __builtin_shufflevector(lo, hi, 0,1,2,3,4,5,6,7,8,9,10,11,12,13,14,15);
}

static __device__ __forceinline__ v8f wmma_bf16(v16bf a, v16bf b, v8f c) {
  // D = A(16x32 bf16) * B(32x16 bf16) + C(16x16 f32)
  return __builtin_amdgcn_wmma_f32_16x16x32_bf16(false, a, false, b, (short)0, c, false, false);
}

// ---------------------------------------------------------------------------
// Kernel 1: Y = X @ W  (X: [B*S, D] f32, W: [D, D] f32) -> bf16, Q scaled 1/32.
// block = 128 threads (4 waves), tile 64(M) x 64(N), K-step 32.
// W tile staged TRANSPOSED in LDS so B-fragments are two 16B reads along k.
// ---------------------------------------------------------------------------
__global__ __launch_bounds__(128) void qkv_gemm(
    const float* __restrict__ x,  const float* __restrict__ Wq,
    const float* __restrict__ Wk, const float* __restrict__ Wv,
    __bf16* __restrict__ Qo, __bf16* __restrict__ Ko, __bf16* __restrict__ Vo)
{
  const int mblk  = blockIdx.x * 64;
  const int nblk  = blockIdx.y * 64;
  const int which = blockIdx.z;
  const float* __restrict__ W   = (which == 0) ? Wq : (which == 1) ? Wk : Wv;
  __bf16*      __restrict__ out = (which == 0) ? Qo : (which == 1) ? Ko : Vo;
  const float oscale = (which == 0) ? 0.03125f : 1.0f;   // 1/sqrt(1024) folded into Q

  // transposed W tile: [n within 64][kd within 32], pitch 40 (80B, 16B multiple)
  __shared__ alignas(16) __bf16 wt[64][40];

  const int tid  = threadIdx.x;
  const int wave = tid >> 5;
  const int lane = tid & 31;
  const int half = lane >> 4;
  const int l16  = lane & 15;

  const float* __restrict__ arow = x + (size_t)(mblk + wave * 16 + l16) * DM;

  v8f acc[4];
#pragma unroll
  for (int s = 0; s < 4; ++s) acc[s] = (v8f){0.f,0.f,0.f,0.f,0.f,0.f,0.f,0.f};

  for (int kb = 0; kb < DM; kb += 32) {
    if (kb + 32 < DM)
      __builtin_prefetch(&W[(size_t)(kb + 32) * DM + nblk], 0, 1);

    // stage 32x64 f32 W tile into LDS transposed (coalesced float4 global reads)
#pragma unroll
    for (int it = 0; it < 4; ++it) {
      int slot = tid + it * 128;             // 512 float4 slots
      int r = slot >> 4;                     // kd 0..31
      int c = (slot & 15) << 2;              // n  0..60
      float4 f = *(const float4*)&W[(size_t)(kb + r) * DM + nblk + c];
      wt[c + 0][r] = (__bf16)f.x;
      wt[c + 1][r] = (__bf16)f.y;
      wt[c + 2][r] = (__bf16)f.z;
      wt[c + 3][r] = (__bf16)f.w;
    }
    __syncthreads();

    // A fragment (row = l16): lanes 0-15 take K 0-7 & 16-23, lanes 16-31 take 8-15 & 24-31
    const int d0 = kb + half * 8;
    float4 xa = *(const float4*)&arow[d0];
    float4 xb = *(const float4*)&arow[d0 + 4];
    float4 xc = *(const float4*)&arow[d0 + 16];
    float4 xd = *(const float4*)&arow[d0 + 20];
    v16bf a;
    a[0]=(__bf16)xa.x; a[1]=(__bf16)xa.y; a[2]=(__bf16)xa.z;  a[3]=(__bf16)xa.w;
    a[4]=(__bf16)xb.x; a[5]=(__bf16)xb.y; a[6]=(__bf16)xb.z;  a[7]=(__bf16)xb.w;
    a[8]=(__bf16)xc.x; a[9]=(__bf16)xc.y; a[10]=(__bf16)xc.z; a[11]=(__bf16)xc.w;
    a[12]=(__bf16)xd.x;a[13]=(__bf16)xd.y;a[14]=(__bf16)xd.z; a[15]=(__bf16)xd.w;

#pragma unroll
    for (int s = 0; s < 4; ++s) {
      // B fragment: lane = col, elem e -> K = 8*half + e (e<8), 16 + 8*half + (e-8) (e>=8) => two 16B reads in wt
      v16bf bf = cat8(*(const v8bf*)&wt[s * 16 + l16][8 * half], *(const v8bf*)&wt[s * 16 + l16][16 + 8 * half]);
      acc[s] = wmma_bf16(a, bf, acc[s]);
    }
    __syncthreads();
  }

  // C/D layout: VGPR r -> row r + 8*half, col = l16
#pragma unroll
  for (int s = 0; s < 4; ++s)
#pragma unroll
    for (int r = 0; r < 8; ++r) {
      const int orow = mblk + wave * 16 + r + 8 * half;
      const int ocol = nblk + s * 16 + l16;
      out[(size_t)orow * DM + ocol] = (__bf16)(acc[s][r] * oscale);
    }
}

// ---------------------------------------------------------------------------
// Kernel 1b: V^T (per batch [S][D] -> [D][S]) via 64x64 LDS tiles, both sides
// coalesced 16B accesses. Makes V B-fragments contiguous 32B global loads.
// ---------------------------------------------------------------------------
__global__ __launch_bounds__(256) void transpose_v(
    const __bf16* __restrict__ Vin, __bf16* __restrict__ Vt)
{
  const int kt = blockIdx.x * 64;   // key tile
  const int et = blockIdx.y * 64;   // e tile
  const int b  = blockIdx.z;
  const __bf16* __restrict__ src = Vin + (size_t)b * SEQ * DM;
  __bf16*       __restrict__ dst = Vt  + (size_t)b * DM * SEQ;

  __shared__ alignas(16) __bf16 tile[64][72];

  const int tid = threadIdx.x;
#pragma unroll
  for (int it = 0; it < 2; ++it) {
    int slot = tid + it * 256;          // 512 uint4 slots
    int r = slot >> 3;                  // key row 0..63
    int c = (slot & 7) * 8;             // e col 0..56
    uint4 u = *(const uint4*)&src[(size_t)(kt + r) * DM + et + c];
    *(uint4*)&tile[r][c] = u;
  }
  __syncthreads();
#pragma unroll
  for (int it = 0; it < 2; ++it) {
    int slot = tid + it * 256;
    int e  = slot >> 3;                 // e row 0..63
    int k0 = (slot & 7) * 8;            // key col 0..56
    __bf16 tmp[8];
#pragma unroll
    for (int j = 0; j < 8; ++j) tmp[j] = tile[k0 + j][e];
    *(uint4*)&dst[(size_t)(et + e) * SEQ + kt + k0] = *(const uint4*)tmp;
  }
}

// ---------------------------------------------------------------------------
// Kernel 2: causal flash attention. One block per (batch, 16-query tile).
// 8 waves; per 256-key step: wave w computes scores for keys [w*32, w*32+32),
// cooperative online softmax through LDS, then each wave accumulates its own
// 128 output columns of O += P @ V (V read transposed: contiguous fragments).
// ---------------------------------------------------------------------------
__global__ __launch_bounds__(256) void flash_attn(
    const __bf16* __restrict__ Q, const __bf16* __restrict__ K,
    const __bf16* __restrict__ Vt, float* __restrict__ out)
{
  const int qt = blockIdx.x;          // 16-row query tile
  const int b  = blockIdx.y;

  const __bf16* __restrict__ Qb  = Q  + (size_t)b * SEQ * DM;
  const __bf16* __restrict__ Kb  = K  + (size_t)b * SEQ * DM;
  const __bf16* __restrict__ Vtb = Vt + (size_t)b * DM * SEQ;

  __shared__ alignas(16) float  s_lds[16][264];   // raw scores for 256 keys
  __shared__ alignas(16) __bf16 p_lds[16][264];   // exp'd probabilities (bf16)
  __shared__ float  red_max[16][16];
  __shared__ float  red_sum[16][16];
  __shared__ float  row_scale[16];
  __shared__ float  row_newm[16];
  __shared__ float  m_run[16];
  __shared__ float  l_run[16];

  const int tid  = threadIdx.x;
  const int wave = tid >> 5;          // 0..7
  const int lane = tid & 31;
  const int half = lane >> 4;
  const int l16  = lane & 15;

  if (tid < 16) { m_run[tid] = -3.0e38f; l_run[tid] = 0.0f; }

  v8f o[8];
#pragma unroll
  for (int s = 0; s < 8; ++s) o[s] = (v8f){0.f,0.f,0.f,0.f,0.f,0.f,0.f,0.f};

  const __bf16* __restrict__ qrow = Qb + (size_t)(qt * 16 + l16) * DM;
  const int nouter = (qt * 16 + 16 + 255) >> 8;   // 256-key blocks up to diagonal

  __syncthreads();

  for (int ko = 0; ko < nouter; ++ko) {
    const int kbase = ko * 256 + wave * 32;

    // ---- S = (Q/sqrt(d)) K^T for this wave's 32 keys, full D sweep ----
    v8f sc0 = (v8f){0.f,0.f,0.f,0.f,0.f,0.f,0.f,0.f};
    v8f sc1 = (v8f){0.f,0.f,0.f,0.f,0.f,0.f,0.f,0.f};
    const __bf16* __restrict__ krow0 = Kb + (size_t)(kbase + l16) * DM + half * 16;
    const __bf16* __restrict__ krow1 = krow0 + (size_t)16 * DM;
    for (int db = 0; db < DM; db += 32) {
      v8bf qlo = *(const v8bf*)&qrow[db + half * 16];
      v8bf qhi = *(const v8bf*)&qrow[db + half * 16 + 8];
      v16bf aq  = cat8(qlo, qhi);
      v16bf bk0 = *(const v16bf*)&krow0[db];    // contiguous 32B B-fragment
      v16bf bk1 = *(const v16bf*)&krow1[db];
      sc0 = wmma_bf16(aq, bk0, sc0);
      sc1 = wmma_bf16(aq, bk1, sc1);
    }

    // ---- causal mask + spill scores ----
#pragma unroll
    for (int r = 0; r < 8; ++r) {
      const int qg = qt * 16 + r + 8 * half;
      float v0 = ((kbase + l16)      > qg) ? -3.0e38f : sc0[r];
      float v1 = ((kbase + 16 + l16) > qg) ? -3.0e38f : sc1[r];
      s_lds[r + 8 * half][wave * 32 + l16]      = v0;
      s_lds[r + 8 * half][wave * 32 + 16 + l16] = v1;
    }
    __syncthreads();

    // ---- per-16-key-chunk max/sum: thread = (row = tid&15, chunk = tid>>4) ----
    {
      const int rr = tid & 15, rc = tid >> 4;
      float4 c0 = *(const float4*)&s_lds[rr][rc * 16 + 0];
      float4 c1 = *(const float4*)&s_lds[rr][rc * 16 + 4];
      float4 c2 = *(const float4*)&s_lds[rr][rc * 16 + 8];
      float4 c3 = *(const float4*)&s_lds[rr][rc * 16 + 12];
      float mx = fmaxf(fmaxf(fmaxf(c0.x, c0.y), fmaxf(c0.z, c0.w)),
                       fmaxf(fmaxf(c1.x, c1.y), fmaxf(c1.z, c1.w)));
      mx = fmaxf(mx, fmaxf(fmaxf(fmaxf(c2.x, c2.y), fmaxf(c2.z, c2.w)),
                           fmaxf(fmaxf(c3.x, c3.y), fmaxf(c3.z, c3.w))));
      float se = __expf(c0.x - mx) + __expf(c0.y - mx) + __expf(c0.z - mx) + __expf(c0.w - mx)
               + __expf(c1.x - mx) + __expf(c1.y - mx) + __expf(c1.z - mx) + __expf(c1.w - mx)
               + __expf(c2.x - mx) + __expf(c2.y - mx) + __expf(c2.z - mx) + __expf(c2.w - mx)
               + __expf(c3.x - mx) + __expf(c3.y - mx) + __expf(c3.z - mx) + __expf(c3.w - mx);
      red_max[rr][rc] = mx;
      red_sum[rr][rc] = se;
    }
    __syncthreads();

    // ---- online-softmax state update (one thread per row) ----
    if (tid < 16) {
      float gmax = m_run[tid];
#pragma unroll
      for (int c = 0; c < 16; ++c) gmax = fmaxf(gmax, red_max[tid][c]);
      float gsum = 0.f;
#pragma unroll
      for (int c = 0; c < 16; ++c) gsum += red_sum[tid][c] * __expf(red_max[tid][c] - gmax);
      const float rs = __expf(m_run[tid] - gmax);
      l_run[tid]     = l_run[tid] * rs + gsum;
      m_run[tid]     = gmax;
      row_scale[tid] = rs;
      row_newm[tid]  = gmax;
    }
    __syncthreads();

    // ---- P = exp(S - m_new) -> bf16 (one 32B vector store) ; rescale O ----
    {
      const int rr = tid & 15, rc = tid >> 4;
      const float nm = row_newm[rr];
      float4 c0 = *(const float4*)&s_lds[rr][rc * 16 + 0];
      float4 c1 = *(const float4*)&s_lds[rr][rc * 16 + 4];
      float4 c2 = *(const float4*)&s_lds[rr][rc * 16 + 8];
      float4 c3 = *(const float4*)&s_lds[rr][rc * 16 + 12];
      v16bf pv;
      pv[0]=(__bf16)__expf(c0.x-nm); pv[1]=(__bf16)__expf(c0.y-nm);
      pv[2]=(__bf16)__expf(c0.z-nm); pv[3]=(__bf16)__expf(c0.w-nm);
      pv[4]=(__bf16)__expf(c1.x-nm); pv[5]=(__bf16)__expf(c1.y-nm);
      pv[6]=(__bf16)__expf(c1.z-nm); pv[7]=(__bf16)__expf(c1.w-nm);
      pv[8]=(__bf16)__expf(c2.x-nm); pv[9]=(__bf16)__expf(c2.y-nm);
      pv[10]=(__bf16)__expf(c2.z-nm); pv[11]=(__bf16)__expf(c2.w-nm);
      pv[12]=(__bf16)__expf(c3.x-nm); pv[13]=(__bf16)__expf(c3.y-nm);
      pv[14]=(__bf16)__expf(c3.z-nm); pv[15]=(__bf16)__expf(c3.w-nm);
      *(v16bf*)&p_lds[rr][rc * 16] = pv;
    }
    {
      float rs[8];
#pragma unroll
      for (int r = 0; r < 8; ++r) rs[r] = row_scale[r + 8 * half];
#pragma unroll
      for (int s = 0; s < 8; ++s)
#pragma unroll
        for (int r = 0; r < 8; ++r) o[s][r] *= rs[r];
    }
    __syncthreads();

    // ---- O += P @ V ; this wave owns e-columns [wave*128, wave*128+128) ----
#pragma unroll 1
    for (int kc = 0; kc < 8; ++kc) {
      // A-fragment of P: two contiguous 16B runs per lane
      v8bf plo = *(const v8bf*)&p_lds[l16][kc * 32 + 16 * half];
      v8bf phi = *(const v8bf*)&p_lds[l16][kc * 32 + 16 * half + 8];
      v16bf pa = cat8(plo, phi);
      const int keyg = ko * 256 + kc * 32 + 16 * half;
#pragma unroll
      for (int s = 0; s < 8; ++s) {
        const int ecol = wave * 128 + s * 16 + l16;
        // B-fragment of V from V^T: contiguous 32B (L2-resident)
        v16bf vf = *(const v16bf*)&Vtb[(size_t)ecol * SEQ + keyg];
        o[s] = wmma_bf16(pa, vf, o[s]);
      }
    }
    __syncthreads();
  }

  // ---- normalize and store fp32 output ----
  float invl[8];
#pragma unroll
  for (int r = 0; r < 8; ++r) invl[r] = 1.0f / l_run[r + 8 * half];
#pragma unroll
  for (int s = 0; s < 8; ++s)
#pragma unroll
    for (int r = 0; r < 8; ++r) {
      const size_t orow = (size_t)b * SEQ + qt * 16 + r + 8 * half;
      out[orow * DM + wave * 128 + s * 16 + l16] = o[s][r] * invl[r];
    }
}

// ---------------------------------------------------------------------------
extern "C" void kernel_launch(void* const* d_in, const int* in_sizes, int n_in,
                              void* d_out, int out_size, void* d_ws, size_t ws_size,
                              hipStream_t stream) {
  const float* x  = (const float*)d_in[0];
  const float* Wq = (const float*)d_in[1];
  const float* Wk = (const float*)d_in[2];
  const float* Wv = (const float*)d_in[3];
  float* out = (float*)d_out;

  const size_t n = (size_t)NB * SEQ * DM;       // 16,777,216 elements
  __bf16* Qw  = (__bf16*)d_ws;                  // 32 MB
  __bf16* Kw  = Qw + n;                         // 32 MB
  __bf16* Vw  = Kw + n;                         // 32 MB
  __bf16* Vtw = Vw + n;                         // 32 MB (V transposed)

  dim3 g1((NB * SEQ) / 64, DM / 64, 3);
  qkv_gemm<<<g1, 128, 0, stream>>>(x, Wq, Wk, Wv, Qw, Kw, Vw);

  dim3 gt(SEQ / 64, DM / 64, NB);
  transpose_v<<<gt, 256, 0, stream>>>(Vw, Vtw);

  dim3 g2(SEQ / 16, NB);
  flash_attn<<<g2, 256, 0, stream>>>(Qw, Kw, Vtw, out);
}
